// EdgeAwareAttention_43344809951969
// MI455X (gfx1250) — matrix operand order repaired and checked by reading, not compiled, not run on hardware
//
#include <hip/hip_runtime.h>
#include <hip/hip_bf16.h>
#include <stdint.h>

#define DD 256
#define HH 8
#define HD 32

typedef __attribute__((ext_vector_type(16))) __bf16 v16bf;
typedef __attribute__((ext_vector_type(8)))  float  v8f;

struct __align__(16) U128 { unsigned int a, b, c, d; };
union FragAB { v16bf v; U128 q[2]; };

__device__ __forceinline__ unsigned short f2bf(float f) {
    unsigned int u = __float_as_uint(f);
    unsigned int r = u + 0x7FFFu + ((u >> 16) & 1u);   // round-to-nearest-even
    return (unsigned short)(r >> 16);
}

// ---------------- elementwise helpers ----------------

__global__ void k_cvt_bf16(const float* __restrict__ in, unsigned short* __restrict__ out, int n) {
    int i = blockIdx.x * blockDim.x + threadIdx.x;
    if (i < n) out[i] = f2bf(in[i]);
}

// Wt[n*D + k] = bf16(W[k*D + n])  (transpose so WMMA B-fragments are contiguous in K)
__global__ void k_transpose_bf16(const float* __restrict__ W, unsigned short* __restrict__ Wt) {
    int i = blockIdx.x * blockDim.x + threadIdx.x;
    if (i < DD * DD) {
        int k = i / DD, n = i % DD;
        Wt[n * DD + k] = f2bf(W[k * DD + n]);
    }
}

__global__ void k_fill_i32(int* __restrict__ p, int v, int n) {
    int i = blockIdx.x * blockDim.x + threadIdx.x;
    if (i < n) p[i] = v;
}

__global__ void k_copy_i32(const int* __restrict__ in, int* __restrict__ out, int n) {
    int i = blockIdx.x * blockDim.x + threadIdx.x;
    if (i < n) out[i] = in[i];
}

// ---------------- WMMA bf16 GEMM: C[M,Ncols] = A[M,K] * B[K,Ncols] ----------------
// A: row-major bf16 [M x Kdim]; Bt: B transposed, row-major bf16 [Ncols x Kdim].
// Block = 256 threads = 8 waves. Wave w -> rows [blk*128 + 16w, +16), 4 N-tiles of 16.
__global__ __launch_bounds__(256)
void k_gemm_bf16_wmma(const unsigned short* __restrict__ A,
                      const unsigned short* __restrict__ Bt,
                      float* __restrict__ C,
                      int M, int Kdim, int Ncols) {
    const int wave = threadIdx.x >> 5;
    const int lane = threadIdx.x & 31;
    const int hs   = lane >> 4;     // half-select: lanes 16-31 hold the shifted K-ranges
    const int r    = lane & 15;
    const int m0   = blockIdx.x * 128 + wave * 16;
    const int n0   = blockIdx.y * 64;

    int rowA = m0 + r; if (rowA >= M) rowA = M - 1;           // clamp (EXEC must stay all-1s for WMMA)
    const unsigned short* Arow  = A  + (size_t)rowA * Kdim;
    const unsigned short* Brow0 = Bt + (size_t)(n0 +  0 + r) * Kdim;
    const unsigned short* Brow1 = Bt + (size_t)(n0 + 16 + r) * Kdim;
    const unsigned short* Brow2 = Bt + (size_t)(n0 + 32 + r) * Kdim;
    const unsigned short* Brow3 = Bt + (size_t)(n0 + 48 + r) * Kdim;

    v8f zero = {0.f,0.f,0.f,0.f,0.f,0.f,0.f,0.f};
    v8f acc0 = zero, acc1 = zero, acc2 = zero, acc3 = zero;

    for (int k0 = 0; k0 < Kdim; k0 += 32) {
        __builtin_prefetch(Arow + k0 + 32, 0, 3);  // global_prefetch next K-slab
        // A 16x32 bf16 fragment: lane holds row m=r, K = 16*hs + e (contiguous, matches B)
        FragAB a;
        a.q[0] = *reinterpret_cast<const U128*>(Arow + k0 + 16 * hs);
        a.q[1] = *reinterpret_cast<const U128*>(Arow + k0 + 16 * hs + 8);
        // B 32x16 bf16 fragment: lane holds column n=r, K = 16*hs + e (contiguous in Bt row)
        const int kb = k0 + 16 * hs;
        FragAB b;
        b.q[0] = *reinterpret_cast<const U128*>(Brow0 + kb);
        b.q[1] = *reinterpret_cast<const U128*>(Brow0 + kb + 8);
        acc0 = __builtin_amdgcn_wmma_f32_16x16x32_bf16(false, a.v, false, b.v, (short)0, acc0, false, false);
        b.q[0] = *reinterpret_cast<const U128*>(Brow1 + kb);
        b.q[1] = *reinterpret_cast<const U128*>(Brow1 + kb + 8);
        acc1 = __builtin_amdgcn_wmma_f32_16x16x32_bf16(false, a.v, false, b.v, (short)0, acc1, false, false);
        b.q[0] = *reinterpret_cast<const U128*>(Brow2 + kb);
        b.q[1] = *reinterpret_cast<const U128*>(Brow2 + kb + 8);
        acc2 = __builtin_amdgcn_wmma_f32_16x16x32_bf16(false, a.v, false, b.v, (short)0, acc2, false, false);
        b.q[0] = *reinterpret_cast<const U128*>(Brow3 + kb);
        b.q[1] = *reinterpret_cast<const U128*>(Brow3 + kb + 8);
        acc3 = __builtin_amdgcn_wmma_f32_16x16x32_bf16(false, a.v, false, b.v, (short)0, acc3, false, false);
    }

    // C/D layout: VGPR j, lane -> M = j + 8*hs, N = r
    const int col0 = n0 + r;
#pragma unroll
    for (int j = 0; j < 8; ++j) {
        int rowc = m0 + 8 * hs + j;
        if (rowc < M) {
            size_t base = (size_t)rowc * Ncols;
            C[base + col0]      = acc0[j];
            C[base + col0 + 16] = acc1[j];
            C[base + col0 + 32] = acc2[j];
            C[base + col0 + 48] = acc3[j];
        }
    }
}

// ---------------- CSR construction (counting sort by src) ----------------

__global__ void k_hist(const int* __restrict__ src, int* __restrict__ cnt, int E) {
    int i = blockIdx.x * blockDim.x + threadIdx.x;
    if (i < E) atomicAdd(&cnt[src[i]], 1);
}

// one-block two-level exclusive scan: off[0..N], off[N] = total
__global__ __launch_bounds__(256)
void k_scan(const int* __restrict__ cnt, int* __restrict__ off, int N) {
    __shared__ int ssum[256];
    __shared__ int sofs[257];
    const int tid = threadIdx.x;
    const int chunk = (N + 255) / 256;
    const int beg = tid * chunk;
    int end = beg + chunk; if (end > N) end = N;
    int s = 0;
    for (int i = beg; i < end; ++i) s += cnt[i];
    ssum[tid] = s;
    __syncthreads();
    if (tid == 0) {
        int run = 0;
        for (int i = 0; i < 256; ++i) { sofs[i] = run; run += ssum[i]; }
        sofs[256] = run;
    }
    __syncthreads();
    int run = sofs[tid];
    for (int i = beg; i < end; ++i) { off[i] = run; run += cnt[i]; }
    if (tid == 255) off[N] = sofs[256];
}

__global__ void k_reorder(const int* __restrict__ src, int* __restrict__ cur,
                          int* __restrict__ perm, int E) {
    int i = blockIdx.x * blockDim.x + threadIdx.x;
    if (i < E) {
        int pos = atomicAdd(&cur[src[i]], 1);
        perm[pos] = i;
    }
}

// ---------------- fused gather attention (online softmax, no atomics) ----------------
// One wave per node. Lane l owns dims [8l, 8l+8); head h = l>>2; per-head dot reduced
// with 2 xor-shuffles across the 4-lane group. Output written directly as bf16 (GEMM A input).
__global__ __launch_bounds__(256)
void k_attn_gather(const float* __restrict__ Q, const float* __restrict__ K,
                   const float* __restrict__ V,
                   const int* __restrict__ off, const int* __restrict__ perm,
                   const int* __restrict__ tgt, const float* __restrict__ ew,
                   const float* __restrict__ We,
                   unsigned short* __restrict__ aggbf, int N) {
    const int wave = threadIdx.x >> 5;
    const int lane = threadIdx.x & 31;
    const int n = blockIdx.x * 8 + wave;
    if (n >= N) return;

    const int h = lane >> 2;
    const float weh = We[h];
    const int base = lane * 8;            // == h*32 + (lane&3)*8

    const float4* qp = reinterpret_cast<const float4*>(Q + (size_t)n * DD + base);
    const float4 q0 = qp[0], q1 = qp[1];

    const int jb = off[n], je = off[n + 1];

    float mx = -3.0e38f;
    float s  = 0.f;
    float4 acc0 = {0.f,0.f,0.f,0.f}, acc1 = {0.f,0.f,0.f,0.f};

    for (int j = jb; j < je; ++j) {
        const int e = perm[j];
        const int t = tgt[e];
        const float4* kp = reinterpret_cast<const float4*>(K + (size_t)t * DD + base);
        const float4 k0 = kp[0], k1 = kp[1];
        float p = q0.x*k0.x + q0.y*k0.y + q0.z*k0.z + q0.w*k0.w
                + q1.x*k1.x + q1.y*k1.y + q1.z*k1.z + q1.w*k1.w;
        p += __shfl_xor(p, 1, 32);        // reduce hd=32 across the 4 lanes of this head
        p += __shfl_xor(p, 2, 32);
        const float sc = p * 0.17677669529663687f + ew[e] * weh;   // 1/sqrt(32)

        // online softmax rescale (first iter: exp(-3e38 - nm) underflows to 0)
        const float nm   = fmaxf(mx, sc);
        const float corr = __expf(mx - nm);
        const float pe   = __expf(sc - nm);
        s = s * corr + pe;

        const float4* vp = reinterpret_cast<const float4*>(V + (size_t)t * DD + base);
        const float4 v0 = vp[0], v1 = vp[1];
        acc0.x = acc0.x * corr + pe * v0.x;
        acc0.y = acc0.y * corr + pe * v0.y;
        acc0.z = acc0.z * corr + pe * v0.z;
        acc0.w = acc0.w * corr + pe * v0.w;
        acc1.x = acc1.x * corr + pe * v1.x;
        acc1.y = acc1.y * corr + pe * v1.y;
        acc1.z = acc1.z * corr + pe * v1.z;
        acc1.w = acc1.w * corr + pe * v1.w;
        mx = nm;
    }

    const float inv = (s > 0.f) ? 1.f / s : 0.f;
    U128 o;
    o.a = (unsigned)f2bf(acc0.x * inv) | ((unsigned)f2bf(acc0.y * inv) << 16);
    o.b = (unsigned)f2bf(acc0.z * inv) | ((unsigned)f2bf(acc0.w * inv) << 16);
    o.c = (unsigned)f2bf(acc1.x * inv) | ((unsigned)f2bf(acc1.y * inv) << 16);
    o.d = (unsigned)f2bf(acc1.z * inv) | ((unsigned)f2bf(acc1.w * inv) << 16);
    *reinterpret_cast<U128*>(aggbf + (size_t)n * DD + base) = o;  // 16B packed bf16 store
}

// ---------------- residual + LayerNorm ----------------

__global__ __launch_bounds__(256)
void k_ln(const float* __restrict__ proj, const float* __restrict__ X,
          const float* __restrict__ g, const float* __restrict__ b,
          float* __restrict__ out, int N) {
    int row = blockIdx.x, tid = threadIdx.x;
    float x = proj[(size_t)row * DD + tid] + X[(size_t)row * DD + tid];
    __shared__ float s[256];
    s[tid] = x; __syncthreads();
    for (int st = 128; st > 0; st >>= 1) { if (tid < st) s[tid] += s[tid + st]; __syncthreads(); }
    float mu = s[0] * (1.f / DD);
    __syncthreads();
    float d = x - mu;
    s[tid] = d * d; __syncthreads();
    for (int st = 128; st > 0; st >>= 1) { if (tid < st) s[tid] += s[tid + st]; __syncthreads(); }
    float var = s[0] * (1.f / DD);
    out[(size_t)row * DD + tid] = d * rsqrtf(var + 1e-5f) * g[tid] + b[tid];
}

// ---------------- launcher ----------------

extern "C" void kernel_launch(void* const* d_in, const int* in_sizes, int n_in,
                              void* d_out, int out_size, void* d_ws, size_t ws_size,
                              hipStream_t stream) {
    const float* X   = (const float*)d_in[0];
    const int*   ei  = (const int*)d_in[1];
    const float* ew  = (const float*)d_in[2];
    const float* Wq  = (const float*)d_in[3];
    const float* Wk  = (const float*)d_in[4];
    const float* Wv  = (const float*)d_in[5];
    const float* We  = (const float*)d_in[6];
    const float* Wo  = (const float*)d_in[7];
    const float* lng = (const float*)d_in[8];
    const float* lnb = (const float*)d_in[9];
    float* out = (float*)d_out;

    const int N  = in_sizes[0] / DD;
    const int E  = in_sizes[1] / 2;
    const int ND = N * DD;
    const int* srcI = ei;
    const int* tgtI = ei + E;

    char* base = (char*)d_ws;
    size_t off_b = 0;
    auto carve = [&](size_t bytes) -> void* {
        void* p = base + off_b;
        off_b += (bytes + 255) & ~(size_t)255;
        return p;
    };
    unsigned short* Xbf = (unsigned short*)carve((size_t)ND * 2);  // later reused as agg-bf16
    unsigned short* Wqt = (unsigned short*)carve((size_t)DD * DD * 2);
    unsigned short* Wkt = (unsigned short*)carve((size_t)DD * DD * 2);
    unsigned short* Wvt = (unsigned short*)carve((size_t)DD * DD * 2);
    unsigned short* Wot = (unsigned short*)carve((size_t)DD * DD * 2);
    float* Q    = (float*)carve((size_t)ND * 4);
    float* Kp   = (float*)carve((size_t)ND * 4);
    float* V    = (float*)carve((size_t)ND * 4);
    int* cnt    = (int*)carve((size_t)N * 4);
    int* offs   = (int*)carve((size_t)(N + 1) * 4);
    int* cur    = (int*)carve((size_t)N * 4);
    int* perm   = (int*)carve((size_t)E * 4);
    float* proj = (float*)carve((size_t)ND * 4);

    dim3 blk(256);
    // 1) precision conversion + weight transposes
    k_cvt_bf16<<<(ND + 255) / 256, blk, 0, stream>>>(X, Xbf, ND);
    k_transpose_bf16<<<(DD * DD) / 256, blk, 0, stream>>>(Wq, Wqt);
    k_transpose_bf16<<<(DD * DD) / 256, blk, 0, stream>>>(Wk, Wkt);
    k_transpose_bf16<<<(DD * DD) / 256, blk, 0, stream>>>(Wv, Wvt);
    k_transpose_bf16<<<(DD * DD) / 256, blk, 0, stream>>>(Wo, Wot);

    // 2) Q/K/V projections (WMMA bf16 GEMMs)
    dim3 ggrid((N + 127) / 128, DD / 64);
    k_gemm_bf16_wmma<<<ggrid, blk, 0, stream>>>(Xbf, Wqt, Q,  N, DD, DD);
    k_gemm_bf16_wmma<<<ggrid, blk, 0, stream>>>(Xbf, Wkt, Kp, N, DD, DD);
    k_gemm_bf16_wmma<<<ggrid, blk, 0, stream>>>(Xbf, Wvt, V,  N, DD, DD);

    // 3) CSR by src (counting sort), then fused gather attention
    k_fill_i32<<<(N + 255) / 256, blk, 0, stream>>>(cnt, 0, N);
    k_hist<<<(E + 255) / 256, blk, 0, stream>>>(srcI, cnt, E);
    k_scan<<<1, blk, 0, stream>>>(cnt, offs, N);
    k_copy_i32<<<(N + 255) / 256, blk, 0, stream>>>(offs, cur, N);
    k_reorder<<<(E + 255) / 256, blk, 0, stream>>>(srcI, cur, perm, E);
    k_attn_gather<<<(N + 7) / 8, blk, 0, stream>>>(Q, Kp, V, offs, perm, tgtI, ew, We,
                                                   /*aggbf=*/Xbf, N);

    // 4) output projection (WMMA) + residual + LayerNorm
    k_gemm_bf16_wmma<<<ggrid, blk, 0, stream>>>(Xbf, Wot, proj, N, DD, DD);
    k_ln<<<N, blk, 0, stream>>>(proj, X, lng, lnb, out, N);
}
